// Decoder_81346680586941
// MI455X (gfx1250) — matrix operand order repaired and checked by reading, not compiled, not run on hardware
//
#include <hip/hip_runtime.h>

// ---------------------------------------------------------------------------
// CfC (3-layer NCP) forward on gfx1250: bf16 WMMA GEMMs with fused activation.
// ---------------------------------------------------------------------------

typedef __attribute__((ext_vector_type(16))) __bf16 v16bf;
typedef __attribute__((ext_vector_type(8)))  float  v8f;
typedef __attribute__((ext_vector_type(4)))  unsigned int u32x4;
typedef __attribute__((ext_vector_type(4)))  float f32x4;

union Frag16 {
  v16bf v;
  u32x4 q[2];
};

__device__ __forceinline__ unsigned short f2bf(float f) {
  union { float f; unsigned int u; } c; c.f = f;
  unsigned int u = c.u;
  u += 0x7FFFu + ((u >> 16) & 1u);   // round-to-nearest-even
  return (unsigned short)(u >> 16);
}

__device__ __forceinline__ v8f wmma_bf16(Frag16 a, Frag16 b, v8f c) {
  return __builtin_amdgcn_wmma_f32_16x16x32_bf16(
      /*neg_a=*/false, a.v, /*neg_b=*/false, b.v,
      /*c_mod=*/(short)0, c, /*reuse_a=*/false, /*reuse_b=*/false);
}

// ---------------------------------------------------------------------------
// One CfC cell step for one layer: hnew = tanh(xc@W1m+b1)*(1-s) + s*tanh(xc@W2m+b2)
// with s = sigmoid(xc@Wa+ba + xc@Wb+bb); xc = [input, h] handled as 2 K-segments.
// Block: 256 threads (8 waves). Tile: 128 rows (batch) x 16 cols (neurons).
// Wave w computes rows [mBase + 16w, +16) for all 4 weight matrices.
// ---------------------------------------------------------------------------
__global__ __launch_bounds__(256) void cfc_cell(
    const void* __restrict__ A0, int a0f32, long lda0, int K0,        // input segment
    const unsigned short* __restrict__ A1, long lda1, int K1,         // recurrent segment (bf16)
    const unsigned short* __restrict__ W, int Ktot, int Npad,         // [4][Npad][Ktot] bf16
    const float* __restrict__ b1, const float* __restrict__ b2,
    const float* __restrict__ b3, const float* __restrict__ b4,
    int Nact,
    unsigned short* __restrict__ hout,                                // (B, Npad) bf16
    float* __restrict__ moutP, long mstride,                          // motor output (or null)
    float* __restrict__ hnP, int writeHn)                             // final hidden slice
{
  // A tile: 128 rows x 32 K halves, row stride 40 halves (80B) -> conflict-free frags
  __shared__ __attribute__((aligned(128))) unsigned short sA[128 * 40];

  const int tid   = threadIdx.x;
  const int lane  = tid & 31;
  const int wave  = tid >> 5;
  const int r16   = lane & 15;   // A: row-in-tile / B,C,D: column
  const int khalf = lane >> 4;
  const int nTile = blockIdx.x * 16;
  const int mBase = blockIdx.y * 128;

  const int ldrow  = tid >> 1;   // cooperative A loader: 2 threads per row
  const int ldpart = tid & 1;

  const long matStride = (long)Npad * Ktot;

  v8f acc0 = {}; v8f acc1 = {}; v8f acc2 = {}; v8f acc3 = {};

  for (int seg = 0; seg < 2; ++seg) {
    const int Klen = seg ? K1 : K0;
    const int wkb  = seg ? K0 : 0;
    for (int kb = 0; kb < Klen; kb += 32) {
      __syncthreads();
      // ---- stage A tile (128 x 32) into LDS as bf16 ----
      {
        unsigned int wpk[8];
        if (!seg && a0f32) {
          const float* src = (const float*)A0 + (long)(mBase + ldrow) * lda0 + kb + ldpart * 16;
          const f32x4* s4 = (const f32x4*)src;
          #pragma unroll
          for (int i = 0; i < 4; ++i) {
            f32x4 fv = s4[i];
            wpk[2*i]   = (unsigned)f2bf(fv[0]) | ((unsigned)f2bf(fv[1]) << 16);
            wpk[2*i+1] = (unsigned)f2bf(fv[2]) | ((unsigned)f2bf(fv[3]) << 16);
          }
        } else {
          const unsigned short* src = (seg ? A1 : (const unsigned short*)A0)
                                    + (long)(mBase + ldrow) * (seg ? lda1 : lda0)
                                    + kb + ldpart * 16;
          const u32x4* s4 = (const u32x4*)src;
          u32x4 u0 = s4[0], u1 = s4[1];
          wpk[0]=u0[0]; wpk[1]=u0[1]; wpk[2]=u0[2]; wpk[3]=u0[3];
          wpk[4]=u1[0]; wpk[5]=u1[1]; wpk[6]=u1[2]; wpk[7]=u1[3];
        }
        u32x4* dst = (u32x4*)&sA[ldrow * 40 + ldpart * 16];
        dst[0] = u32x4{wpk[0], wpk[1], wpk[2], wpk[3]};
        dst[1] = u32x4{wpk[4], wpk[5], wpk[6], wpk[7]};
      }
      __syncthreads();

      // ---- A fragment (16x32 bf16 per-lane layout) from LDS ----
      Frag16 af;
      {
        const unsigned short* ap = &sA[(wave * 16 + r16) * 40 + khalf * 16];
        af.q[0] = *(const u32x4*)ap;        // K = khalf*16 .. +7
        af.q[1] = *(const u32x4*)(ap + 8);  // K = khalf*16 + 8 .. +7
      }
      // ---- B fragments straight from L2 (W rows contiguous in K) ----
      const unsigned short* bbase = W + (long)(nTile + r16) * Ktot + wkb + kb + khalf * 16;
      {
        Frag16 bf;
        bf.q[0] = *(const u32x4*)bbase;  bf.q[1] = *(const u32x4*)(bbase + 8);
        acc0 = wmma_bf16(af, bf, acc0);
      }
      {
        Frag16 bf;
        const unsigned short* bp = bbase + matStride;
        bf.q[0] = *(const u32x4*)bp;     bf.q[1] = *(const u32x4*)(bp + 8);
        acc1 = wmma_bf16(af, bf, acc1);
      }
      {
        Frag16 bf;
        const unsigned short* bp = bbase + 2 * matStride;
        bf.q[0] = *(const u32x4*)bp;     bf.q[1] = *(const u32x4*)(bp + 8);
        acc2 = wmma_bf16(af, bf, acc2);
      }
      {
        Frag16 bf;
        const unsigned short* bp = bbase + 3 * matStride;
        bf.q[0] = *(const u32x4*)bp;     bf.q[1] = *(const u32x4*)(bp + 8);
        acc3 = wmma_bf16(af, bf, acc3);
      }
    }
  }

  // ---- fused CfC activation epilogue ----
  const int nco = nTile + r16;
  const float bb1  = (nco < Nact) ? b1[nco] : 0.f;
  const float bb2  = (nco < Nact) ? b2[nco] : 0.f;
  const float bb34 = (nco < Nact) ? (b3[nco] + b4[nco]) : 0.f;
  const int rbase = mBase + wave * 16 + (khalf ? 8 : 0);
  #pragma unroll
  for (int r = 0; r < 8; ++r) {
    const long m = rbase + r;
    const float ff1 = tanhf(acc0[r] + bb1);
    const float ff2 = tanhf(acc1[r] + bb2);
    const float ts  = acc2[r] + acc3[r] + bb34;
    const float s   = 1.f / (1.f + expf(-ts));
    const float h   = ff1 * (1.f - s) + s * ff2;
    hout[m * Npad + nco] = f2bf(h);
    if (moutP)                 moutP[m * mstride + nco] = h;
    if (writeHn && nco < Nact) hnP[m * 1024 + nco] = h;
  }
}

// ---------------------------------------------------------------------------
// Final FC: predictions = out @ Wfc^T + bfc, in place over (B*T, 128) rows.
// Each block owns 128 rows entirely (reads them to LDS before writing).
// ---------------------------------------------------------------------------
__global__ __launch_bounds__(256) void fc_wmma(
    float* __restrict__ data, const unsigned short* __restrict__ WfcBf,
    const float* __restrict__ bfc)
{
  __shared__ __attribute__((aligned(128))) unsigned short sA[128 * 136];
  const int tid  = threadIdx.x;
  const int lane = tid & 31, wave = tid >> 5;
  const int r16  = lane & 15, khalf = lane >> 4;
  const long rowBase = (long)blockIdx.x * 128;

  {
    const int row = tid >> 1, part = tid & 1;
    const float* src = data + (rowBase + row) * 128 + part * 64;
    unsigned short* d = &sA[row * 136 + part * 64];
    #pragma unroll
    for (int i = 0; i < 8; ++i) {
      const f32x4* s4 = (const f32x4*)(src + i * 8);
      f32x4 a = s4[0], b = s4[1];
      u32x4 w;
      w[0] = (unsigned)f2bf(a[0]) | ((unsigned)f2bf(a[1]) << 16);
      w[1] = (unsigned)f2bf(a[2]) | ((unsigned)f2bf(a[3]) << 16);
      w[2] = (unsigned)f2bf(b[0]) | ((unsigned)f2bf(b[1]) << 16);
      w[3] = (unsigned)f2bf(b[2]) | ((unsigned)f2bf(b[3]) << 16);
      *(u32x4*)(d + i * 8) = w;
    }
  }
  __syncthreads();

  v8f acc[8];
  #pragma unroll
  for (int nt = 0; nt < 8; ++nt) acc[nt] = (v8f){};

  for (int kb = 0; kb < 128; kb += 32) {
    Frag16 af;
    const unsigned short* ap = &sA[(wave * 16 + r16) * 136 + kb + khalf * 16];
    af.q[0] = *(const u32x4*)ap;
    af.q[1] = *(const u32x4*)(ap + 8);
    #pragma unroll
    for (int nt = 0; nt < 8; ++nt) {
      Frag16 bf;
      const unsigned short* bp = WfcBf + (nt * 16 + r16) * 128 + kb + khalf * 16;
      bf.q[0] = *(const u32x4*)bp;
      bf.q[1] = *(const u32x4*)(bp + 8);
      acc[nt] = wmma_bf16(af, bf, acc[nt]);
    }
  }

  const int rbase = wave * 16 + (khalf ? 8 : 0);
  #pragma unroll
  for (int nt = 0; nt < 8; ++nt) {
    const int n = nt * 16 + r16;
    const float bias = bfc[n];
    #pragma unroll
    for (int r = 0; r < 8; ++r)
      data[(rowBase + rbase + r) * 128 + n] = acc[nt][r] + bias;
  }
}

// ---------------------------------------------------------------------------
// Prep: masked weights -> bf16 [4][Npad][Ktot]; h0 -> padded bf16; Wfc -> bf16.
// ---------------------------------------------------------------------------
__global__ void prep_w(const float* __restrict__ W1, const float* __restrict__ W2,
                       const float* __restrict__ Wa, const float* __restrict__ Wb,
                       const float* __restrict__ mask,
                       int Nact, int inTot, int Kx, int Kxpad, int Ktot, int Npad,
                       unsigned short* __restrict__ dst, long total)
{
  const long perMat = (long)Npad * Ktot;
  for (long i = blockIdx.x * (long)blockDim.x + threadIdx.x; i < total;
       i += (long)gridDim.x * blockDim.x) {
    const int mat  = (int)(i / perMat);
    const long rem = i - (long)mat * perMat;
    const int np = (int)(rem / Ktot);
    const int kp = (int)(rem - (long)np * Ktot);
    float v = 0.f;
    if (np < Nact) {
      int ks = -1;
      if (kp < Kxpad) { if (kp < Kx) ks = kp; }
      else { const int kh = kp - Kxpad; if (kh < Nact) ks = Kx + kh; }
      if (ks >= 0) {
        const float* Wsrc = (mat == 0) ? W1 : (mat == 1) ? W2 : (mat == 2) ? Wa : Wb;
        v = Wsrc[(long)np * inTot + ks];
        if (mat < 2) v *= mask[(long)np * inTot + ks];
      }
    }
    dst[i] = f2bf(v);
  }
}

__global__ void prep_h(const float* __restrict__ h0, int hoff, int Nact, int Npad,
                       unsigned short* __restrict__ dst, long total)
{
  for (long i = blockIdx.x * (long)blockDim.x + threadIdx.x; i < total;
       i += (long)gridDim.x * blockDim.x) {
    const int b = (int)(i / Npad);
    const int n = (int)(i - (long)b * Npad);
    const float v = (n < Nact) ? h0[(long)b * 1024 + hoff + n] : 0.f;
    dst[i] = f2bf(v);
  }
}

__global__ void prep_cvt(const float* __restrict__ src, unsigned short* __restrict__ dst,
                         long total)
{
  for (long i = blockIdx.x * (long)blockDim.x + threadIdx.x; i < total;
       i += (long)gridDim.x * blockDim.x)
    dst[i] = f2bf(src[i]);
}

// ---------------------------------------------------------------------------
extern "C" void kernel_launch(void* const* d_in, const int* in_sizes, int n_in,
                              void* d_out, int out_size, void* d_ws, size_t ws_size,
                              hipStream_t stream) {
  (void)in_sizes; (void)n_in; (void)out_size; (void)ws_size;

  const float* x  = (const float*)d_in[0];
  const float* h0 = (const float*)d_in[1];
  const float *mask[3], *Wff1[3], *Wff2[3], *Wta[3], *Wtb[3];
  const float *bff1[3], *bff2[3], *bta[3], *btb[3];
  for (int l = 0; l < 3; ++l) {
    const int b = 2 + l * 9;
    mask[l] = (const float*)d_in[b + 0];
    Wff1[l] = (const float*)d_in[b + 1];
    Wff2[l] = (const float*)d_in[b + 2];
    Wta[l]  = (const float*)d_in[b + 3];
    Wtb[l]  = (const float*)d_in[b + 4];
    bff1[l] = (const float*)d_in[b + 5];
    bff2[l] = (const float*)d_in[b + 6];
    bta[l]  = (const float*)d_in[b + 7];
    btb[l]  = (const float*)d_in[b + 8];
  }
  const float* Wfc = (const float*)d_in[29];
  const float* bfc = (const float*)d_in[30];

  static const int Nact[3]  = {538, 358, 128};
  static const int inTot[3] = {1050, 896, 486};
  static const int Kx[3]    = {512, 538, 358};
  static const int Kxpad[3] = {512, 544, 384};   // Kxpad[l] == Npad[l-1] (or 512 for x)
  static const int Npad[3]  = {544, 384, 128};   // multiples of 32
  static const int Ktot[3]  = {1056, 928, 512};
  static const int hoff[3]  = {0, 538, 896};

  // workspace carve (bf16 units), ~10.2 MB total
  unsigned short* wsb = (unsigned short*)d_ws;
  const long szW[3] = {4L * 544 * 1056, 4L * 384 * 928, 4L * 128 * 512};
  const long szH[3] = {512L * 544, 512L * 384, 512L * 128};
  unsigned short* Wall[3];
  long off = 0;
  for (int l = 0; l < 3; ++l) { Wall[l] = wsb + off; off += szW[l]; }
  unsigned short* WfcBf = wsb + off; off += 128L * 128;
  unsigned short* hbuf[3];
  for (int l = 0; l < 3; ++l) { hbuf[l] = wsb + off; off += 2 * szH[l]; }

  // ---- prep ----
  for (int l = 0; l < 3; ++l)
    prep_w<<<(int)((szW[l] + 255) / 256), 256, 0, stream>>>(
        Wff1[l], Wff2[l], Wta[l], Wtb[l], mask[l],
        Nact[l], inTot[l], Kx[l], Kxpad[l], Ktot[l], Npad[l], Wall[l], szW[l]);
  for (int l = 0; l < 3; ++l)
    prep_h<<<(int)((szH[l] + 255) / 256), 256, 0, stream>>>(
        h0, hoff[l], Nact[l], Npad[l], hbuf[l], szH[l]);
  prep_cvt<<<64, 256, 0, stream>>>(Wfc, WfcBf, 128L * 128);

  float* pred = (float*)d_out;                    // (B, T, 128) staging then predictions
  float* hn   = pred + 512L * 128 * 128;          // (B, 1024)

  // ---- sequential CfC scan ----
  for (int t = 0; t < 128; ++t) {
    const int rp = t & 1, wp = 1 - rp;
    const int last = (t == 127) ? 1 : 0;
    for (int l = 0; l < 3; ++l) {
      const void* A0; int a0f32; long lda0;
      if (l == 0) { A0 = (const void*)(x + (long)t * 512); a0f32 = 1; lda0 = 128L * 512; }
      else        { A0 = (const void*)(hbuf[l - 1] + (long)wp * szH[l - 1]); a0f32 = 0; lda0 = Npad[l - 1]; }
      const unsigned short* A1 = hbuf[l] + (long)rp * szH[l];
      unsigned short* hout     = hbuf[l] + (long)wp * szH[l];
      float* moutP = (l == 2) ? (pred + (long)t * 128) : nullptr;
      dim3 grid(Npad[l] / 16, 4);
      cfc_cell<<<grid, 256, 0, stream>>>(
          A0, a0f32, lda0, Kxpad[l],
          A1, (long)Npad[l], Npad[l],
          Wall[l], Ktot[l], Npad[l],
          bff1[l], bff2[l], bta[l], btb[l], Nact[l],
          hout, moutP, 128L * 128,
          hn + hoff[l], last);
    }
  }

  // ---- final linear layer (in place over prediction region) ----
  fc_wmma<<<512, 256, 0, stream>>>(pred, WfcBf, bfc);
}
